// SelfAttention_2491081032412
// MI455X (gfx1250) — compile-verified, matrix operand order checked by reading, not run on hardware
//
#include <hip/hip_runtime.h>
#include <hip/hip_bf16.h>

// ---------------------------------------------------------------------------
// Causal self-attention with per-head LayerNorm (kqnorm), MI455X / gfx1250.
// BF16 WMMA (v_wmma_f32_16x16x32_bf16) with f32 accumulation everywhere.
// Projections: 4-wave blocks, weights staged into LDS by the Tensor Data
// Mover (tensor_load_to_lds + s_wait_tensorcnt), double-buffered.
// Attention: 1-wave flash-attention tiles (EXEC all-ones at every WMMA).
// ---------------------------------------------------------------------------

#define EMB       1024
#define HEADS     16
#define HEAD_DIM  64
#define BATCH     2
#define SEQ       2048
#define MTILES    ((BATCH * SEQ) / 16)   // 256
#define LN_EPS    1e-5f
#define INV4      0.17677669529663687f   // 1024^(-0.25)

typedef __attribute__((ext_vector_type(16))) __bf16 v16bf;
typedef __attribute__((ext_vector_type(8)))  float  v8f;
typedef __attribute__((ext_vector_type(4)))  unsigned int v4u;
typedef __attribute__((ext_vector_type(8)))  int    v8i;
typedef __attribute__((ext_vector_type(4)))  int    v4i;

__device__ __forceinline__ v8f vzero8() {
    return (v8f){0.f, 0.f, 0.f, 0.f, 0.f, 0.f, 0.f, 0.f};
}

// D = A*B + C, 16x16x32 bf16 -> f32
__device__ __forceinline__ v8f wmma_bf16(v16bf a, v16bf b, v8f c) {
    return __builtin_amdgcn_wmma_f32_16x16x32_bf16(
        /*neg_a=*/false, a, /*neg_b=*/false, b,
        /*c_mod=*/(short)0, c, /*reuse_a=*/false, /*reuse_b=*/false);
}

// Load a 16x32 bf16 fragment (A-layout per CDNA5 ISA 7.12.2; B-fragments use
// the same per-lane pattern with lane = column index).  Row-major source,
// row stride `ld` elements.  Two contiguous 16-byte loads per lane.
__device__ __forceinline__ v16bf load_frag(const __bf16* p, int ld) {
    const int lane = threadIdx.x & 31;
    const int r    = lane & 15;
    const int koff = (lane & 16) ? 8 : 0;
    const __bf16* q = p + (size_t)r * (size_t)ld + koff;
    v16bf f;
    *reinterpret_cast<uint4*>(&f)       = *reinterpret_cast<const uint4*>(q);
    *(reinterpret_cast<uint4*>(&f) + 1) = *reinterpret_cast<const uint4*>(q + 16);
    return f;
}

// ---------------------------------------------------------------------------
// TDM: DMA a 2D bf16 tile (tile_d0 contiguous elems x tile_d1 rows, row
// stride `stride0` elems) from global memory into LDS (packed row-major).
// D# layout per CDNA5 ISA ch.8: group0 = {count/lds_addr/global_addr/type},
// group1 = {data_size, tensor dims, tile dims, stride}.  Groups 2/3 zeroed
// (2D tensor).  Issued by one wave; completion via TENSORcnt.
// ---------------------------------------------------------------------------
__device__ __forceinline__ void tdm_load_tile_bf16(void* lds_dst, const void* gsrc,
                                                   unsigned tensor_d0, unsigned tensor_d1,
                                                   unsigned stride0,
                                                   unsigned tile_d0, unsigned tile_d1) {
    const unsigned lds_off = (unsigned)(uintptr_t)lds_dst;     // low 32 bits = LDS offset
    const unsigned long long ga = (unsigned long long)(uintptr_t)gsrc;
    v4u g0;
    g0[0] = 1u;                                                // count=1 user descriptor
    g0[1] = lds_off;                                           // lds_addr [63:32]
    g0[2] = (unsigned)(ga & 0xffffffffu);                      // global_addr[31:0]
    g0[3] = 0x80000000u | (unsigned)((ga >> 32) & 0x01ffffffu); // addr[56:32] | type=2
    v8i g1;
    g1[0] = (int)(1u << 16);                                   // data_size=1 (2 bytes)
    g1[1] = (int)(tensor_d0 << 16);                            // tensor_dim0[15:0]
    g1[2] = (int)((tensor_d0 >> 16) | (tensor_d1 << 16));      // dim0 hi | dim1 lo
    g1[3] = (int)((tensor_d1 >> 16) | (tile_d0 << 16));        // dim1 hi | tile_dim0
    g1[4] = (int)(tile_d1 & 0xffffu);                          // tile_dim1 (tile_dim2=0)
    g1[5] = (int)stride0;                                      // tensor_dim0_stride lo
    g1[6] = 0;                                                 // stride0 hi | stride1 lo
    g1[7] = 0;
    v4i gz = {0, 0, 0, 0};
#if __clang_major__ >= 23
    v8i gz8 = {0, 0, 0, 0, 0, 0, 0, 0};
    __builtin_amdgcn_tensor_load_to_lds(g0, g1, gz, gz, gz8, 0);
#else
    __builtin_amdgcn_tensor_load_to_lds(g0, g1, gz, gz, 0);
#endif
}

// ---------------------------------------------------------------------------
// Stage 0: f32 -> bf16 conversion
// ---------------------------------------------------------------------------
__global__ void cvt_f32_bf16(const float* __restrict__ src,
                             __bf16* __restrict__ dst, int n) {
    int i = blockIdx.x * blockDim.x + threadIdx.x;
    if (i < n) dst[i] = (__bf16)src[i];
}

// ---------------------------------------------------------------------------
// Stage 1: QKV projection + per-head LayerNorm + emb^-0.25 scaling.
// grid = (MTILES/4, HEADS, 3)  z: 0=Q, 1=K, 2=V.   block = 128 (4 waves).
// Block computes 64 rows x 64 cols (one head).  Weight tile (64 x k-chunk 64)
// is TDM-staged into LDS, double-buffered, shared by all 4 waves.  Wave w
// owns rows 16w..16w+15.  Outputs:
//   Q,K -> head-major [b, h, t, 64] (LayerNorm + INV4)
//   V   -> transposed [b, h, 64, t]
// ---------------------------------------------------------------------------
__global__ void qkv_ln_kernel(const __bf16* __restrict__ xh,
                              const __bf16* __restrict__ Wqh,
                              const __bf16* __restrict__ Wkh,
                              const __bf16* __restrict__ Wvh,
                              const float*  __restrict__ qg,
                              const float*  __restrict__ qb,
                              const float*  __restrict__ kg,
                              const float*  __restrict__ kb,
                              __bf16* __restrict__ Qh,
                              __bf16* __restrict__ Kh,
                              __bf16* __restrict__ Vt) {
    const int mt64 = blockIdx.x;     // 64-row tile of flattened (B*T)
    const int h    = blockIdx.y;
    const int z    = blockIdx.z;
    const int w    = threadIdx.x >> 5;
    const int lane = threadIdx.x & 31;

    const __bf16* W = (z == 0) ? Wqh : ((z == 1) ? Wkh : Wvh);
    const __bf16* wbase = W + (size_t)(h * HEAD_DIM) * EMB;      // 64 weight rows
    const __bf16* arow  = xh + (size_t)(mt64 * 64 + w * 16) * EMB;

    __shared__ __attribute__((aligned(16))) __bf16 Bs[2][64 * 64];   // 16 KB
    __shared__ __attribute__((aligned(16))) float  tile[4][16 * 64]; // 16 KB

    if (w == 0) {
        tdm_load_tile_bf16(Bs[0], wbase,      EMB, EMB, EMB, 64, 64);
        tdm_load_tile_bf16(Bs[1], wbase + 64, EMB, EMB, EMB, 64, 64);
    }

    v8f acc[4] = {vzero8(), vzero8(), vzero8(), vzero8()};
    for (int kc = 0; kc < 16; ++kc) {                // 16 chunks of K=64
        if (w == 0) {
            if (kc == 15) __builtin_amdgcn_s_wait_tensorcnt(0);
            else          __builtin_amdgcn_s_wait_tensorcnt(1);
        }
        __syncthreads();
        const __bf16* bs = Bs[kc & 1];
        const int k0 = kc * 64;
#pragma unroll
        for (int ks = 0; ks < 64; ks += 32) {
            v16bf a = load_frag(arow + k0 + ks, EMB);
#pragma unroll
            for (int nt = 0; nt < 4; ++nt) {
                v16bf b = load_frag(bs + (nt * 16) * 64 + ks, 64);  // ds_load frags
                acc[nt] = wmma_bf16(a, b, acc[nt]);
            }
        }
        __syncthreads();                              // everyone done with buf
        if (w == 0 && kc + 2 < 16)
            tdm_load_tile_bf16(Bs[kc & 1], wbase + (kc + 2) * 64,
                               EMB, EMB, EMB, 64, 64);
    }

    // C-layout -> per-wave LDS tile (16 rows x 64 cols, f32)
#pragma unroll
    for (int nt = 0; nt < 4; ++nt)
#pragma unroll
        for (int r = 0; r < 8; ++r) {
            int m = r + ((lane & 16) ? 8 : 0);
            tile[w][m * 64 + nt * 16 + (lane & 15)] = acc[nt][r];
        }
    __syncthreads();

    const int row0 = mt64 * 64 + w * 16;             // flattened (b*T + t)
    const int bb   = row0 / SEQ;
    const int tloc = row0 % SEQ;                     // never crosses batch

    if (z < 2) {
        if (lane < 16) {
            const float* g   = (z == 0) ? qg : kg;
            const float* be  = (z == 0) ? qb : kb;
            const float* row = tile[w] + lane * 64;
            float mu = 0.f;
#pragma unroll
            for (int j = 0; j < 64; ++j) mu += row[j];
            mu *= (1.f / 64.f);
            float var = 0.f;
#pragma unroll
            for (int j = 0; j < 64; ++j) { float d = row[j] - mu; var += d * d; }
            var *= (1.f / 64.f);
            const float inv = rsqrtf(var + LN_EPS) * INV4;   // fold emb^-0.25
            __bf16* dst = ((z == 0) ? Qh : Kh) +
                (((size_t)(bb * HEADS + h)) * SEQ + (size_t)(tloc + lane)) * HEAD_DIM;
#pragma unroll
            for (int j = 0; j < 64; ++j)
                dst[j] = (__bf16)((row[j] - mu) * inv * g[j] + be[j] * INV4);
        }
    } else {
        // V: store transposed [b, h, s, t]
        __bf16* dstv = Vt + ((size_t)(bb * HEADS + h)) * HEAD_DIM * SEQ + tloc;
#pragma unroll
        for (int ss = 0; ss < 2; ++ss) {
            int s = lane * 2 + ss;
#pragma unroll
            for (int i = 0; i < 16; ++i)
                dstv[(size_t)s * SEQ + i] = (__bf16)tile[w][i * 64 + s];
        }
    }
}

// ---------------------------------------------------------------------------
// Stage 2: causal flash attention.  grid = (SEQ/16, HEADS, BATCH), block = 32.
//   S(16x32) = Q(16x64) @ K^T  : 2 accumulators x 2 WMMAs
//   O(16x64) += P(16x32) @ V   : P routed through LDS (C-layout -> A-layout)
// ---------------------------------------------------------------------------
__global__ void attn_kernel(const __bf16* __restrict__ Qh,
                            const __bf16* __restrict__ Kh,
                            const __bf16* __restrict__ Vt,
                            __bf16* __restrict__ aout) {
    const int qt   = blockIdx.x;
    const int h    = blockIdx.y;
    const int bb   = blockIdx.z;
    const int lane = threadIdx.x;
    const float NEG_INF = -__builtin_inff();

    const size_t bh = (size_t)(bb * HEADS + h);
    const __bf16* Qbase = Qh + (bh * SEQ + (size_t)qt * 16) * HEAD_DIM;
    const __bf16* Kbase = Kh + bh * SEQ * HEAD_DIM;
    const __bf16* Vbase = Vt + bh * HEAD_DIM * SEQ;

    const v16bf aq0 = load_frag(Qbase,      HEAD_DIM);   // K = 0..31
    const v16bf aq1 = load_frag(Qbase + 32, HEAD_DIM);   // K = 32..63

    v8f o[4] = {vzero8(), vzero8(), vzero8(), vzero8()};
    float mrow[8], lrow[8];
#pragma unroll
    for (int r = 0; r < 8; ++r) { mrow[r] = NEG_INF; lrow[r] = 0.f; }

    __shared__ __attribute__((aligned(16))) __bf16 Plds[16 * 32];

    const int kmax = qt * 16 + 15;                  // last valid key (causal)
    for (int kt = 0; kt <= kmax; kt += 32) {
        v8f s[2];
#pragma unroll
        for (int g = 0; g < 2; ++g) {
            v16bf bk0 = load_frag(Kbase + (size_t)(kt + g * 16) * HEAD_DIM,      HEAD_DIM);
            v16bf bk1 = load_frag(Kbase + (size_t)(kt + g * 16) * HEAD_DIM + 32, HEAD_DIM);
            v8f acc = vzero8();
            acc = wmma_bf16(aq0, bk0, acc);
            acc = wmma_bf16(aq1, bk1, acc);
            const int n_global = kt + g * 16 + (lane & 15);
#pragma unroll
            for (int r = 0; r < 8; ++r) {
                int m_global = qt * 16 + r + ((lane & 16) ? 8 : 0);
                if (n_global > m_global) acc[r] = NEG_INF;
            }
            s[g] = acc;
        }

#pragma unroll
        for (int r = 0; r < 8; ++r) {
            float tm = fmaxf(s[0][r], s[1][r]);
            tm = fmaxf(tm, __shfl_xor(tm, 1, 32));
            tm = fmaxf(tm, __shfl_xor(tm, 2, 32));
            tm = fmaxf(tm, __shfl_xor(tm, 4, 32));
            tm = fmaxf(tm, __shfl_xor(tm, 8, 32));
            const float nm = fmaxf(mrow[r], tm);
            const float sc = __expf(mrow[r] - nm);
            const float p0 = __expf(s[0][r] - nm);  // masked -> exp(-inf)=0
            const float p1 = __expf(s[1][r] - nm);
            float rs = p0 + p1;
            rs += __shfl_xor(rs, 1, 32);
            rs += __shfl_xor(rs, 2, 32);
            rs += __shfl_xor(rs, 4, 32);
            rs += __shfl_xor(rs, 8, 32);
            lrow[r] = lrow[r] * sc + rs;
            mrow[r] = nm;
            o[0][r] *= sc; o[1][r] *= sc; o[2][r] *= sc; o[3][r] *= sc;
            const int m = r + ((lane & 16) ? 8 : 0);
            Plds[m * 32 +      (lane & 15)] = (__bf16)p0;
            Plds[m * 32 + 16 + (lane & 15)] = (__bf16)p1;
        }
        __syncthreads();

        const v16bf pa = load_frag(Plds, 32);       // ds_load fragment
#pragma unroll
        for (int st = 0; st < 4; ++st) {
            v16bf vb = load_frag(Vbase + (size_t)(st * 16) * SEQ + kt, SEQ);
            o[st] = wmma_bf16(pa, vb, o[st]);
        }
        __syncthreads();
    }

    __bf16* dst = aout + ((size_t)bb * SEQ + (size_t)qt * 16) * EMB + h * HEAD_DIM;
#pragma unroll
    for (int r = 0; r < 8; ++r) {
        const int m = r + ((lane & 16) ? 8 : 0);
        const float inv = 1.0f / lrow[r];
#pragma unroll
        for (int st = 0; st < 4; ++st)
            dst[(size_t)m * EMB + st * 16 + (lane & 15)] = (__bf16)(o[st][r] * inv);
    }
}

// ---------------------------------------------------------------------------
// Stage 3: output projection  out = aout @ Wu^T + bu   (f32 result)
// grid = (MTILES/4, EMB/64), block = 128 (4 waves).  Same TDM-staged scheme.
// ---------------------------------------------------------------------------
__global__ void outproj_kernel(const __bf16* __restrict__ aout,
                               const __bf16* __restrict__ Wuh,
                               const float*  __restrict__ bu,
                               float* __restrict__ out) {
    const int mt64 = blockIdx.x;
    const int n64  = blockIdx.y;
    const int w    = threadIdx.x >> 5;
    const int lane = threadIdx.x & 31;

    const __bf16* wbase = Wuh + (size_t)(n64 * 64) * EMB;
    const __bf16* arow  = aout + (size_t)(mt64 * 64 + w * 16) * EMB;

    __shared__ __attribute__((aligned(16))) __bf16 Bs[2][64 * 64];   // 16 KB

    if (w == 0) {
        tdm_load_tile_bf16(Bs[0], wbase,      EMB, EMB, EMB, 64, 64);
        tdm_load_tile_bf16(Bs[1], wbase + 64, EMB, EMB, EMB, 64, 64);
    }

    v8f acc[4] = {vzero8(), vzero8(), vzero8(), vzero8()};
    for (int kc = 0; kc < 16; ++kc) {
        if (w == 0) {
            if (kc == 15) __builtin_amdgcn_s_wait_tensorcnt(0);
            else          __builtin_amdgcn_s_wait_tensorcnt(1);
        }
        __syncthreads();
        const __bf16* bs = Bs[kc & 1];
        const int k0 = kc * 64;
#pragma unroll
        for (int ks = 0; ks < 64; ks += 32) {
            v16bf a = load_frag(arow + k0 + ks, EMB);
#pragma unroll
            for (int nt = 0; nt < 4; ++nt) {
                v16bf b = load_frag(bs + (nt * 16) * 64 + ks, 64);
                acc[nt] = wmma_bf16(a, b, acc[nt]);
            }
        }
        __syncthreads();
        if (w == 0 && kc + 2 < 16)
            tdm_load_tile_bf16(Bs[kc & 1], wbase + (kc + 2) * 64,
                               EMB, EMB, EMB, 64, 64);
    }

#pragma unroll
    for (int nt = 0; nt < 4; ++nt)
#pragma unroll
        for (int r = 0; r < 8; ++r) {
            const int m   = r + ((lane & 16) ? 8 : 0);
            const int col = n64 * 64 + nt * 16 + (lane & 15);
            out[((size_t)(mt64 * 64) + w * 16 + m) * EMB + col] = acc[nt][r] + bu[col];
        }
}

// ---------------------------------------------------------------------------
// Host launcher.  Workspace layout (48 MiB total):
//   [ 0,  8M) xh        [ 8M,10M) Wq_bf16   [10M,12M) Wk_bf16
//   [12M,14M) Wv_bf16   [14M,16M) Wu_bf16
//   [16M,24M) Qh [b,h,t,64]   [24M,32M) Kh [b,h,t,64]
//   [32M,40M) Vt [b,h,64,t]   [40M,48M) aout [b,t,1024]
// ---------------------------------------------------------------------------
extern "C" void kernel_launch(void* const* d_in, const int* in_sizes, int n_in,
                              void* d_out, int out_size, void* d_ws, size_t ws_size,
                              hipStream_t stream) {
    const float* x   = (const float*)d_in[0];
    const float* Wk  = (const float*)d_in[1];
    const float* Wq  = (const float*)d_in[2];
    const float* Wv  = (const float*)d_in[3];
    const float* Wu  = (const float*)d_in[4];
    const float* bu  = (const float*)d_in[5];
    const float* kg  = (const float*)d_in[6];
    const float* kb  = (const float*)d_in[7];
    const float* qg  = (const float*)d_in[8];
    const float* qb  = (const float*)d_in[9];
    float* out = (float*)d_out;

    char* ws = (char*)d_ws;
    const size_t MiB = 1u << 20;
    __bf16* xh   = (__bf16*)(ws + 0 * MiB);
    __bf16* Wqh  = (__bf16*)(ws + 8 * MiB);
    __bf16* Wkh  = (__bf16*)(ws + 10 * MiB);
    __bf16* Wvh  = (__bf16*)(ws + 12 * MiB);
    __bf16* Wuh  = (__bf16*)(ws + 14 * MiB);
    __bf16* Qh   = (__bf16*)(ws + 16 * MiB);
    __bf16* Kh   = (__bf16*)(ws + 24 * MiB);
    __bf16* Vt   = (__bf16*)(ws + 32 * MiB);
    __bf16* aout = (__bf16*)(ws + 40 * MiB);

    const int nx = BATCH * SEQ * EMB;   // 4,194,304
    const int nw = EMB * EMB;           // 1,048,576
    cvt_f32_bf16<<<(nx + 255) / 256, 256, 0, stream>>>(x,  xh,  nx);
    cvt_f32_bf16<<<(nw + 255) / 256, 256, 0, stream>>>(Wq, Wqh, nw);
    cvt_f32_bf16<<<(nw + 255) / 256, 256, 0, stream>>>(Wk, Wkh, nw);
    cvt_f32_bf16<<<(nw + 255) / 256, 256, 0, stream>>>(Wv, Wvh, nw);
    cvt_f32_bf16<<<(nw + 255) / 256, 256, 0, stream>>>(Wu, Wuh, nw);

    qkv_ln_kernel<<<dim3(MTILES / 4, HEADS, 3), 128, 0, stream>>>(
        xh, Wqh, Wkh, Wvh, qg, qb, kg, kb, Qh, Kh, Vt);

    attn_kernel<<<dim3(SEQ / 16, HEADS, BATCH), 32, 0, stream>>>(Qh, Kh, Vt, aout);

    outproj_kernel<<<dim3(MTILES / 4, EMB / 64), 128, 0, stream>>>(aout, Wuh, bu, out);
}
